// Block_39659728011532
// MI455X (gfx1250) — matrix operand order repaired and checked by reading, compiled, not run on hardware
//
#include <hip/hip_runtime.h>
#include <hip/hip_bf16.h>

// ---------------------------------------------------------------------------
// Problem constants (match reference)
// ---------------------------------------------------------------------------
constexpr int kB  = 4;
constexpr int kT  = 2048;
constexpr int kC  = 1024;
constexpr int kH  = 16;
constexpr int kD  = 64;
constexpr int kBT = kB * kT;        // 8192 rows
constexpr int kC4 = 4 * kC;         // 4096

typedef float  v8f   __attribute__((ext_vector_type(8)));
typedef __bf16 v16bf __attribute__((ext_vector_type(16)));

// ---------------------------------------------------------------------------
// Helpers
// ---------------------------------------------------------------------------
static __device__ __forceinline__ unsigned short f2bf(float f) {
    unsigned int u = __builtin_bit_cast(unsigned int, f);
    u += 0x7FFFu + ((u >> 16) & 1u);          // round-to-nearest-even
    return (unsigned short)(u >> 16);
}

// Two 16-byte chunks -> one 16-element bf16 fragment (A or B operand).
static __device__ __forceinline__ v16bf load_bf16_frag(const unsigned short* p0,
                                                       const unsigned short* p1) {
    union { uint4 q[2]; v16bf v; } t;
    t.q[0] = *reinterpret_cast<const uint4*>(p0);
    t.q[1] = *reinterpret_cast<const uint4*>(p1);
    return t.v;
}

static __device__ __forceinline__ v8f wmma_bf16(v16bf a, v16bf b, v8f c) {
    // D = A(16x32 bf16) * B(32x16 bf16) + C(16x16 f32)
    return __builtin_amdgcn_wmma_f32_16x16x32_bf16(
        /*neg_a=*/false, a, /*neg_b=*/false, b,
        /*c_mod=*/(short)0, c, /*reuse_a=*/false, /*reuse_b=*/false);
}

// ---------------------------------------------------------------------------
// Weight convert + transpose: src fp32 [batch][R][Cc]  ->  dst bf16 [batch][Cc][R]
// Produces the [N][K] column-major-B layout every WMMA B-fragment wants.
// ---------------------------------------------------------------------------
__global__ void cvtT_kernel(const float* __restrict__ src,
                            unsigned short* __restrict__ dst,
                            int R, int Ccols, long total) {
    long idx = (long)blockIdx.x * blockDim.x + threadIdx.x;
    if (idx >= total) return;
    long plane = (long)R * Ccols;
    long bm  = idx / plane;
    long rem = idx - bm * plane;
    int  r   = (int)(rem / Ccols);
    int  c   = (int)(rem % Ccols);
    dst[bm * plane + (long)c * R + r] = f2bf(src[idx]);
}

// ---------------------------------------------------------------------------
// LayerNorm -> bf16 row (one block per row, C = 1024, 256 threads)
// ---------------------------------------------------------------------------
__global__ __launch_bounds__(256)
void ln_bf16_kernel(const float* __restrict__ x,
                    const float* __restrict__ g,
                    const float* __restrict__ be,
                    unsigned short* __restrict__ out) {
    const int row = blockIdx.x;
    const int tid = threadIdx.x;
    const float* xr = x + (size_t)row * kC;

    float v[4], s = 0.f, s2 = 0.f;
#pragma unroll
    for (int i = 0; i < 4; ++i) {
        v[i] = xr[tid + i * 256];
        s  += v[i];
        s2 += v[i] * v[i];
    }
    __shared__ float sh1[256], sh2[256];
    sh1[tid] = s; sh2[tid] = s2;
    __syncthreads();
#pragma unroll
    for (int off = 128; off > 0; off >>= 1) {
        if (tid < off) { sh1[tid] += sh1[tid + off]; sh2[tid] += sh2[tid + off]; }
        __syncthreads();
    }
    const float mean = sh1[0] * (1.f / kC);
    const float var  = sh2[0] * (1.f / kC) - mean * mean;
    const float rstd = rsqrtf(var + 1e-5f);
#pragma unroll
    for (int i = 0; i < 4; ++i) {
        int c = tid + i * 256;
        out[(size_t)row * kC + c] = f2bf((v[i] - mean) * rstd * g[c] + be[c]);
    }
}

// ---------------------------------------------------------------------------
// Generic WMMA GEMM: out[M][N] = A[M][K](bf16) * Bt[N][K](bf16)^T (+bias)(+res)(relu)
// Block: 256 threads = 8 waves, block tile 128x128.
// Wave tile 32x64: 2 resident A-fragments, each B-fragment reused twice
// (6 fragment loads per 8 WMMAs -> higher math/load ratio than 16x64).
// OM: 0 = plain [M][N]; 1 = scatter bf16 to [B,H,T,D]; 2 = scatter bf16 to [B,H,D,T]
// ---------------------------------------------------------------------------
template<int OM, bool BIAS, bool RELU, bool RES, bool F32O, bool B16O>
__global__ __launch_bounds__(256)
void gemm_wmma_kernel(const unsigned short* __restrict__ A,
                      const unsigned short* __restrict__ Bt,
                      const float* __restrict__ bias,
                      const float* __restrict__ res,
                      float* __restrict__ outF,
                      unsigned short* __restrict__ outB,
                      int M, int N, int K) {
    const int lane = threadIdx.x & 31;
    const int wave = threadIdx.x >> 5;
    const int half = lane >> 4;      // 0: lanes 0-15, 1: lanes 16-31
    const int ln   = lane & 15;
    const int m0 = blockIdx.x * 128 + (wave & 3) * 32;
    const int n0 = blockIdx.y * 128 + (wave >> 2) * 64;

    v8f acc[2][4];
#pragma unroll
    for (int i = 0; i < 2; ++i)
#pragma unroll
        for (int a = 0; a < 4; ++a) acc[i][a] = v8f{};

    // A fragments: lane holds row (m0 + i*16 + ln); chunks at k0+(half?8:0) and +16.
    const unsigned short* Arow0 = A + (size_t)(m0 + ln)      * K + (half ? 8 : 0);
    const unsigned short* Arow1 = A + (size_t)(m0 + 16 + ln) * K + (half ? 8 : 0);
    // B fragments: lane holds column (n0+a*16+ln); chunks at k0+(half?8:0) and +16.
    const unsigned short* Bcol[4];
#pragma unroll
    for (int a = 0; a < 4; ++a)
        Bcol[a] = Bt + (size_t)(n0 + a * 16 + ln) * K + (half ? 8 : 0);

    for (int k0 = 0; k0 < K; k0 += 32) {
        if (k0 + 32 < K) {
            __builtin_prefetch(Arow0 + k0 + 32, 0, 1);    // global_prefetch_b8
            __builtin_prefetch(Arow1 + k0 + 32, 0, 1);
            __builtin_prefetch(Bcol[0] + k0 + 32, 0, 1);
        }
        v16bf aF0 = load_bf16_frag(Arow0 + k0, Arow0 + k0 + 16);
        v16bf aF1 = load_bf16_frag(Arow1 + k0, Arow1 + k0 + 16);
#pragma unroll
        for (int a = 0; a < 4; ++a) {
            v16bf bF = load_bf16_frag(Bcol[a] + k0, Bcol[a] + k0 + 16);
            acc[0][a] = wmma_bf16(aF0, bF, acc[0][a]);
            acc[1][a] = wmma_bf16(aF1, bF, acc[1][a]);
        }
    }

    // Epilogue.  C/D layout: vgpr r -> row tilebase + r + 8*half, col n0 + a*16 + ln.
#pragma unroll
    for (int i = 0; i < 2; ++i) {
#pragma unroll
        for (int a = 0; a < 4; ++a) {
#pragma unroll
            for (int r = 0; r < 8; ++r) {
                const int row = m0 + i * 16 + r + half * 8;
                const int col = n0 + a * 16 + ln;
                float v = acc[i][a][r];
                if (BIAS) v += bias[col];
                if (RES)  v += res[(size_t)row * N + col];
                if (RELU) v = v > 0.f ? v : 0.f;
                if (OM == 0) {
                    if (F32O) outF[(size_t)row * N + col] = v;
                    if (B16O) outB[(size_t)row * N + col] = f2bf(v);
                } else {
                    const int bb = row / kT, t = row % kT;
                    const int h  = col >> 6, d = col & 63;
                    if (OM == 1)       // q,k : [B,H,T,D]
                        outB[(((size_t)bb * kH + h) * kT + t) * kD + d] = f2bf(v);
                    else               // v   : [B,H,D,T]  (pre-transposed for P*V)
                        outB[(((size_t)bb * kH + h) * kD + d) * kT + t] = f2bf(v);
                }
            }
        }
    }
}

// ---------------------------------------------------------------------------
// Flash-style causal attention. One wave owns one (b, h, 16-query tile).
// Q,K in [B,H,T,D] bf16; V pre-transposed [B,H,D,T] bf16.
// Scale = C^-0.5 = 1/32 (reference quirk: full embed dim, not head dim).
// ---------------------------------------------------------------------------
__global__ __launch_bounds__(256)
void attn_kernel(const unsigned short* __restrict__ q,
                 const unsigned short* __restrict__ k,
                 const unsigned short* __restrict__ vT,
                 unsigned short* __restrict__ o) {
    __shared__ float ldsP[8][16 * 32];     // per-wave 16x32 probability tile
    const int lane = threadIdx.x & 31;
    const int wave = threadIdx.x >> 5;
    const int half = lane >> 4;
    const int ln   = lane & 15;

    const int wid = blockIdx.x * 8 + wave;
    const int nqt = kT / 16;
    const int qt  = wid % nqt;
    const int bh  = wid / nqt;             // b*H + h
    const int h   = bh % kH;
    const int b   = bh / kH;
    const int m0  = qt * 16;

    const unsigned short* qb  = q  + (size_t)bh * kT * kD;
    const unsigned short* kb  = k  + (size_t)bh * kT * kD;
    const unsigned short* vtb = vT + (size_t)bh * kD * kT;
    float* P = ldsP[wave];

    // Q A-fragments for K-dim = 64 (two k-steps of 32), kept resident.
    v16bf qf[2];
#pragma unroll
    for (int kk = 0; kk < 2; ++kk) {
        const unsigned short* p = qb + (size_t)(m0 + ln) * kD + kk * 32 + half * 8;
        qf[kk] = load_bf16_frag(p, p + 16);
    }

    v8f oacc[4];
#pragma unroll
    for (int a = 0; a < 4; ++a) oacc[a] = v8f{};
    float mi[8], li[8];
#pragma unroll
    for (int r = 0; r < 8; ++r) { mi[r] = -__builtin_inff(); li[r] = 0.f; }

    for (int jb = 0; jb < m0 + 16; jb += 32) {
        // ---- scores: two 16x16 tiles (keys jb..+15, jb+16..+31) ----
        v8f sc[2]; sc[0] = v8f{}; sc[1] = v8f{};
#pragma unroll
        for (int kk = 0; kk < 2; ++kk) {
#pragma unroll
            for (int j = 0; j < 2; ++j) {
                const unsigned short* p =
                    kb + (size_t)(jb + j * 16 + ln) * kD + kk * 32 + half * 8;
                v16bf bF = load_bf16_frag(p, p + 16);
                sc[j] = wmma_bf16(qf[kk], bF, sc[j]);
            }
        }
        // ---- scale + causal mask (fp32, register resident) ----
        float ps[2][8];
#pragma unroll
        for (int j = 0; j < 2; ++j)
#pragma unroll
            for (int r = 0; r < 8; ++r) {
                const int row = m0 + r + half * 8;
                const int key = jb + j * 16 + ln;
                const float sv = sc[j][r] * 0.03125f;   // C^-0.5
                ps[j][r] = (key <= row) ? sv : -__builtin_inff();
            }
        // ---- online softmax (row reductions across 16 lanes per half) ----
#pragma unroll
        for (int r = 0; r < 8; ++r) {
            float t = fmaxf(ps[0][r], ps[1][r]);
            t = fmaxf(t, __shfl_xor(t, 8, 32));
            t = fmaxf(t, __shfl_xor(t, 4, 32));
            t = fmaxf(t, __shfl_xor(t, 2, 32));
            t = fmaxf(t, __shfl_xor(t, 1, 32));
            const float mn = fmaxf(mi[r], t);
            const float al = __expf(mi[r] - mn);
            mi[r] = mn;
            float rs = 0.f;
#pragma unroll
            for (int j = 0; j < 2; ++j) {
                const float e = __expf(ps[j][r] - mn);
                ps[j][r] = e;
                rs += e;
            }
            rs += __shfl_xor(rs, 8, 32);
            rs += __shfl_xor(rs, 4, 32);
            rs += __shfl_xor(rs, 2, 32);
            rs += __shfl_xor(rs, 1, 32);
            li[r] = li[r] * al + rs;
#pragma unroll
            for (int a = 0; a < 4; ++a) oacc[a][r] *= al;
        }
        // ---- C-layout -> A-layout via per-wave LDS round trip ----
#pragma unroll
        for (int j = 0; j < 2; ++j)
#pragma unroll
            for (int r = 0; r < 8; ++r)
                P[(r + half * 8) * 32 + j * 16 + ln] = ps[j][r];
        asm volatile("s_wait_dscnt 0x0" ::: "memory");   // same-wave DS RAW fence

        unsigned short pu[16];
        {
            const float* pr = P + ln * 32 + (half ? 8 : 0);
#pragma unroll
            for (int e = 0; e < 8; ++e) {
                pu[e]     = f2bf(pr[e]);        // K = kA .. kA+7
                pu[8 + e] = f2bf(pr[16 + e]);   // K = kA+16 .. kA+23
            }
        }
        v16bf pF;
        {
            union { unsigned short u[16]; v16bf v; } t;
#pragma unroll
            for (int i = 0; i < 16; ++i) t.u[i] = pu[i];
            pF = t.v;
        }
        // ---- P(16x32) * V(32x16) per 16-wide d tile; V is [D][T] so loads are contiguous
#pragma unroll
        for (int a = 0; a < 4; ++a) {
            const unsigned short* p = vtb + (size_t)(a * 16 + ln) * kT + jb + half * 8;
            v16bf bF = load_bf16_frag(p, p + 16);
            oacc[a] = wmma_bf16(pF, bF, oacc[a]);
        }
    }

    // ---- normalize + write concat-head bf16 output [B,T,C] ----
#pragma unroll
    for (int a = 0; a < 4; ++a)
#pragma unroll
        for (int r = 0; r < 8; ++r) {
            const int row = m0 + r + half * 8;
            const int d   = a * 16 + ln;
            const float val = oacc[a][r] / li[r];
            o[((size_t)b * kT + row) * kC + h * 64 + d] = f2bf(val);
        }
}

// ---------------------------------------------------------------------------
// Host-side orchestration
// ---------------------------------------------------------------------------
extern "C" void kernel_launch(void* const* d_in, const int* in_sizes, int n_in,
                              void* d_out, int out_size, void* d_ws, size_t ws_size,
                              hipStream_t stream) {
    (void)in_sizes; (void)n_in; (void)out_size; (void)ws_size;

    const float* x      = (const float*)d_in[0];
    const float* wq     = (const float*)d_in[1];
    const float* wk     = (const float*)d_in[2];
    const float* wv     = (const float*)d_in[3];
    const float* w_proj = (const float*)d_in[4];
    const float* b_proj = (const float*)d_in[5];
    const float* w1     = (const float*)d_in[6];
    const float* b1     = (const float*)d_in[7];
    const float* w2     = (const float*)d_in[8];
    const float* b2     = (const float*)d_in[9];
    const float* g1     = (const float*)d_in[10];
    const float* be1    = (const float*)d_in[11];
    const float* g2     = (const float*)d_in[12];
    const float* be2    = (const float*)d_in[13];
    float* out = (float*)d_out;

    // Workspace carve-up (256B aligned slabs).
    char* wsp = (char*)d_ws;
    auto alloc = [&](size_t bytes) -> void* {
        void* p = wsp;
        wsp += (bytes + 255) & ~(size_t)255;
        return p;
    };
    unsigned short* xn   = (unsigned short*)alloc((size_t)kBT * kC * 2);
    unsigned short* wqT  = (unsigned short*)alloc((size_t)kC * kC * 2);
    unsigned short* wkT  = (unsigned short*)alloc((size_t)kC * kC * 2);
    unsigned short* wvT  = (unsigned short*)alloc((size_t)kC * kC * 2);
    unsigned short* wpT  = (unsigned short*)alloc((size_t)kC * kC * 2);
    unsigned short* w1T  = (unsigned short*)alloc((size_t)kC * kC4 * 2);
    unsigned short* w2T  = (unsigned short*)alloc((size_t)kC * kC4 * 2);
    unsigned short* qb   = (unsigned short*)alloc((size_t)kBT * kC * 2);
    unsigned short* kb   = (unsigned short*)alloc((size_t)kBT * kC * 2);
    unsigned short* vTb  = (unsigned short*)alloc((size_t)kBT * kC * 2);
    unsigned short* attb = (unsigned short*)alloc((size_t)kBT * kC * 2);
    float*          x1   = (float*)alloc((size_t)kBT * kC * 4);
    unsigned short* xn2  = (unsigned short*)alloc((size_t)kBT * kC * 2);
    unsigned short* hb   = (unsigned short*)alloc((size_t)kBT * kC4 * 2);

    // 1) Weight convert+transpose to bf16 [N][K].
    {
        long tq = (long)kH * kC * kD;                       // per-head CxD -> DxC
        cvtT_kernel<<<dim3((tq + 255) / 256), 256, 0, stream>>>(wq, wqT, kC, kD, tq);
        cvtT_kernel<<<dim3((tq + 255) / 256), 256, 0, stream>>>(wk, wkT, kC, kD, tq);
        cvtT_kernel<<<dim3((tq + 255) / 256), 256, 0, stream>>>(wv, wvT, kC, kD, tq);
        long tp = (long)kC * kC;
        cvtT_kernel<<<dim3((tp + 255) / 256), 256, 0, stream>>>(w_proj, wpT, kC, kC, tp);
        long t1 = (long)kC * kC4;
        cvtT_kernel<<<dim3((t1 + 255) / 256), 256, 0, stream>>>(w1, w1T, kC, kC4, t1);
        cvtT_kernel<<<dim3((t1 + 255) / 256), 256, 0, stream>>>(w2, w2T, kC4, kC, t1);
    }

    // 2) LN1 -> xn (bf16)
    ln_bf16_kernel<<<dim3(kBT), 256, 0, stream>>>(x, g1, be1, xn);

    // 3) Q,K,V projections (WMMA), q/k -> [B,H,T,D], v -> [B,H,D,T]
    {
        dim3 grid(kBT / 128, kC / 128);
        gemm_wmma_kernel<1, false, false, false, false, true>
            <<<grid, 256, 0, stream>>>(xn, wqT, nullptr, nullptr, nullptr, qb, kBT, kC, kC);
        gemm_wmma_kernel<1, false, false, false, false, true>
            <<<grid, 256, 0, stream>>>(xn, wkT, nullptr, nullptr, nullptr, kb, kBT, kC, kC);
        gemm_wmma_kernel<2, false, false, false, false, true>
            <<<grid, 256, 0, stream>>>(xn, wvT, nullptr, nullptr, nullptr, vTb, kBT, kC, kC);
    }

    // 4) Causal attention (flash-style, WMMA for QK^T and P*V)
    {
        const int waves = kB * kH * (kT / 16);              // 8192
        attn_kernel<<<dim3(waves / 8), 256, 0, stream>>>(qb, kb, vTb, attb);
    }

    // 5) Output projection + bias + residual(x) -> x1 (fp32)
    {
        dim3 grid(kBT / 128, kC / 128);
        gemm_wmma_kernel<0, true, false, true, true, false>
            <<<grid, 256, 0, stream>>>(attb, wpT, b_proj, x, x1, nullptr, kBT, kC, kC);
    }

    // 6) LN2 -> xn2 (bf16)
    ln_bf16_kernel<<<dim3(kBT), 256, 0, stream>>>(x1, g2, be2, xn2);

    // 7) FFN up + bias + ReLU -> h (bf16)
    {
        dim3 grid(kBT / 128, kC4 / 128);
        gemm_wmma_kernel<0, true, true, false, false, true>
            <<<grid, 256, 0, stream>>>(xn2, w1T, b1, nullptr, nullptr, hb, kBT, kC4, kC);
    }

    // 8) FFN down + bias + residual(x1) -> out (fp32)
    {
        dim3 grid(kBT / 128, kC / 128);
        gemm_wmma_kernel<0, true, false, true, true, false>
            <<<grid, 256, 0, stream>>>(hb, w2T, b2, x1, out, nullptr, kBT, kC, kC4);
    }
}
